// NearestEmbed_45999099740649
// MI455X (gfx1250) — matrix operand order repaired and checked by reading, not compiled, not run on hardware
//
#include <hip/hip_runtime.h>

typedef __attribute__((ext_vector_type(16))) __bf16 v16bf;
typedef __attribute__((ext_vector_type(8)))  __bf16 v8bf;
typedef __attribute__((ext_vector_type(8)))  float  v8f;

union BF16x16 { v16bf v; v8bf h[2]; };

#define DDIM  256   // embedding dim (GEMM K-reduction)
#define KDIM  512   // number of codes
#define NTOK  1024  // H*W tokens per batch image
#define BDIM  64
#define TM    128   // tokens per workgroup
#define THALF 64    // tokens staged in LDS at a time
#define DPAD  264   // padded LDS row stride (mult of 8 -> 16B-aligned b128 frag reads)

// ---------------------------------------------------------------------------
// Kernel 0: codebook prep.  w (D,K) fp32 -> wT_hi/wT_lo (K,D) bf16 split,
// plus e_sq[k] = sum_d w[d][k]^2 (fp32, exact).
// ---------------------------------------------------------------------------
__global__ __launch_bounds__(256) void k_prep(const float* __restrict__ w,
                                              __bf16* __restrict__ wThi,
                                              __bf16* __restrict__ wTlo,
                                              float* __restrict__ eSq) {
  __shared__ float red[256];
  const int k = blockIdx.x;   // 0..511
  const int d = threadIdx.x;  // 0..255
  const float v = w[(size_t)d * KDIM + k];
  const __bf16 h = (__bf16)v;
  const __bf16 l = (__bf16)(v - (float)h);
  wThi[(size_t)k * DDIM + d] = h;
  wTlo[(size_t)k * DDIM + d] = l;
  red[d] = v * v;
  __syncthreads();
  for (int s = 128; s > 0; s >>= 1) {
    if (d < s) red[d] += red[d + s];
    __syncthreads();
  }
  if (d == 0) eSq[k] = red[0];
}

// ---------------------------------------------------------------------------
// Kernel 1: WMMA GEMM + fused argmin.
// score[n,k] = e_sq[k] - 2 * (x_n . w_k), argmin over k.
// One WG = 128 consecutive tokens of one batch image; each wave owns 16
// tokens and sweeps all K=512 codes, so no inter-wave reduction is needed.
// ---------------------------------------------------------------------------
__global__ __launch_bounds__(256) void k_argmin(const float* __restrict__ x,
                                                const __bf16* __restrict__ wThi,
                                                const __bf16* __restrict__ wTlo,
                                                const float* __restrict__ eSq,
                                                int* __restrict__ kmin,
                                                float* __restrict__ argminF) {
  __shared__ __attribute__((aligned(16))) __bf16 sX[THALF * DPAD];
  __shared__ float sE[KDIM];

  const int tid  = threadIdx.x;
  const int lane = tid & 31;
  const int wave = tid >> 5;        // 0..7
  const int ln   = lane & 15;       // column / M index within fragment
  const int hi16 = lane >> 4;       // 0 or 1 (lane half)

  const int tile = blockIdx.x;      // 0..511
  const int b    = tile >> 3;
  const int n0   = (tile & 7) * TM;
  const float* xb = x + (size_t)b * DDIM * NTOK + n0;

  // stage e_sq into LDS (read once per k-tile later)
  sE[tid]       = eSq[tid];
  sE[tid + 256] = eSq[tid + 256];

  // --- stage X tile through LDS, hoist A fragments (hi & lo) to registers ---
  BF16x16 ah[8], al[8];
  const int myHalf = wave >> 2;                 // which 64-token half this wave owns
  const int tloc   = (wave & 3) * 16 + ln;      // A-frag row (token within half)

  for (int half = 0; half < 2; ++half) {
    for (int part = 0; part < 2; ++part) {
      __syncthreads();  // protect sX from previous readers
      {
        // 256 threads cover 64 tokens x 256 dims for this (half, part)
        const int t     = tid & 63;
        const int dbase = (tid >> 6) * 64;  // 0,64,128,192
        const float* src = xb + half * THALF + t + (size_t)dbase * NTOK;
        unsigned int* dst = (unsigned int*)&sX[t * DPAD + dbase];
#pragma unroll
        for (int j = 0; j < 32; ++j) {
          const float v0 = src[(size_t)(2 * j) * NTOK];
          const float v1 = src[(size_t)(2 * j + 1) * NTOK];
          __bf16 b0, b1;
          if (part == 0) {
            b0 = (__bf16)v0;
            b1 = (__bf16)v1;
          } else {
            const __bf16 h0 = (__bf16)v0, h1 = (__bf16)v1;
            b0 = (__bf16)(v0 - (float)h0);
            b1 = (__bf16)(v1 - (float)h1);
          }
          union { __bf16 bb[2]; unsigned int u; } pk;
          pk.bb[0] = b0; pk.bb[1] = b1;
          dst[j] = pk.u;
        }
      }
      __syncthreads();
      if (myHalf == half) {
        BF16x16* dstf = part ? al : ah;
        // ISA A layout (16-bit 16x32): lane<16 holds K = d0..d0+7 (v0-3) and
        // d0+16..23 (v4-7); lane>=16 holds d0+8..15 and d0+24..31.
#pragma unroll
        for (int dsb = 0; dsb < 8; ++dsb) {
          const __bf16* p = &sX[tloc * DPAD + dsb * 32 + hi16 * 8];
          dstf[dsb].h[0] = *(const v8bf*)p;
          dstf[dsb].h[1] = *(const v8bf*)(p + 16);
        }
      }
    }
  }

  // --- K sweep: 32 tiles of 16 codes, D reduced in 8 steps of 32 ---
  float bestv[8];
  int   bestk[8];
#pragma unroll
  for (int r = 0; r < 8; ++r) { bestv[r] = 3.4e38f; bestk[r] = 0; }

  for (int kt = 0; kt < 32; ++kt) {
    const int k0 = kt * 16;
    v8f c = {0.f, 0.f, 0.f, 0.f, 0.f, 0.f, 0.f, 0.f};
    // ISA B layout (16-bit KxN=32x16): lane holds column n=lane%16;
    // lanes 0-15 hold K=0..7,16..23, lanes 16-31 hold K=8..15,24..31 (same as A)
    // -> two 16B global loads of the wT row per fragment.
    const __bf16* bhp = wThi + (size_t)(k0 + ln) * DDIM + hi16 * 8;
    const __bf16* blp = wTlo + (size_t)(k0 + ln) * DDIM + hi16 * 8;
#pragma unroll
    for (int dsb = 0; dsb < 8; ++dsb) {
      BF16x16 bh, bl;
      bh.h[0] = *(const v8bf*)(bhp + dsb * 32);
      bh.h[1] = *(const v8bf*)(bhp + dsb * 32 + 16);
      bl.h[0] = *(const v8bf*)(blp + dsb * 32);
      bl.h[1] = *(const v8bf*)(blp + dsb * 32 + 16);
      // cross ~= xh*wh + xh*wl + xl*wh  (fp32 accumulate)
      c = __builtin_amdgcn_wmma_f32_16x16x32_bf16(false, ah[dsb].v, false, bh.v,
                                                  (short)0, c, false, false);
      c = __builtin_amdgcn_wmma_f32_16x16x32_bf16(false, ah[dsb].v, false, bl.v,
                                                  (short)0, c, false, false);
      c = __builtin_amdgcn_wmma_f32_16x16x32_bf16(false, al[dsb].v, false, bh.v,
                                                  (short)0, c, false, false);
    }
    const float e    = sE[k0 + ln];
    const int   kidx = k0 + ln;
#pragma unroll
    for (int r = 0; r < 8; ++r) {
      const float s = e - 2.0f * c[r];
      if (s < bestv[r]) { bestv[r] = s; bestk[r] = kidx; }
    }
  }

  // --- reduce (val,idx) across the 16 lanes holding one token row ---
#pragma unroll
  for (int r = 0; r < 8; ++r) {
    float v = bestv[r];
    int   k = bestk[r];
#pragma unroll
    for (int off = 8; off >= 1; off >>= 1) {
      const float ov = __shfl_xor(v, off, 32);
      const int   ok = __shfl_xor(k, off, 32);
      if (ov < v || (ov == v && ok < k)) { v = ov; k = ok; }  // first-index tie-break
    }
    bestv[r] = v;
    bestk[r] = k;
  }
  if (ln == 0) {
    // C layout: this lane-half holds rows m = r + 8*hi16.
    const int wbase = (wave >> 2) * 64 + (wave & 3) * 16;
#pragma unroll
    for (int r = 0; r < 8; ++r) {
      const int t     = wbase + hi16 * 8 + r;
      const int nglob = tile * TM + t;  // == b*1024 + token
      kmin[nglob]    = bestk[r];
      argminF[nglob] = (float)bestk[r];
    }
  }
}

// ---------------------------------------------------------------------------
// Kernel 2: quant gather.  out[b][d][n] = w[d][kmin[b*1024+n]].
// 64 MB of coalesced float4 stores; codebook rows (2 KB) are cache-hot.
// ---------------------------------------------------------------------------
__global__ __launch_bounds__(256) void k_quant(const float* __restrict__ w,
                                               const int* __restrict__ kmin,
                                               float* __restrict__ out) {
  const size_t idx4 = (size_t)blockIdx.x * 256 + threadIdx.x;  // 4 outputs each
  const int n4 = (int)(idx4 & 255);   // 1024/4 groups per (b,d) row
  const int bd = (int)(idx4 >> 8);    // b*256 + d
  const int d  = bd & 255;
  const int b  = bd >> 8;
  const int4 kk = *(const int4*)(kmin + b * NTOK + n4 * 4);
  const float* wr = w + (size_t)d * KDIM;
  float4 o;
  o.x = wr[kk.x];
  o.y = wr[kk.y];
  o.z = wr[kk.z];
  o.w = wr[kk.w];
  *(float4*)(out + (size_t)bd * NTOK + n4 * 4) = o;
}

// ---------------------------------------------------------------------------
extern "C" void kernel_launch(void* const* d_in, const int* in_sizes, int n_in,
                              void* d_out, int out_size, void* d_ws, size_t ws_size,
                              hipStream_t stream) {
  (void)in_sizes; (void)n_in; (void)out_size; (void)ws_size;
  const float* x = (const float*)d_in[0];   // (B, D, H, W) fp32
  const float* w = (const float*)d_in[1];   // (D, K) fp32
  float* out = (float*)d_out;               // quant (B,D,H,W) ++ argmin (B,H,W)

  // workspace layout (needs ~788 KB)
  char* ws = (char*)d_ws;
  __bf16* wThi = (__bf16*)(ws);                       // 512*256*2 = 262144 B
  __bf16* wTlo = (__bf16*)(ws + 262144);              // 262144 B
  float*  eSq  = (float*)(ws + 524288);               // 2048 B
  int*    kmin = (int*)(ws + 526336);                 // 65536*4 = 262144 B

  k_prep<<<KDIM, 256, 0, stream>>>(w, wThi, wTlo, eSq);

  const int nTiles = (BDIM * NTOK) / TM;  // 512
  k_argmin<<<nTiles, 256, 0, stream>>>(x, wThi, wTlo, eSq, kmin,
                                       out + (size_t)BDIM * DDIM * NTOK);

  const int nQuant = (BDIM * DDIM * NTOK) / 4 / 256;  // 16384
  k_quant<<<nQuant, 256, 0, stream>>>(w, kmin, out);
}
